// XCA_Token_1700807050103
// MI455X (gfx1250) — matrix operand order repaired and checked by reading, not compiled, not run on hardware
//
#include <hip/hip_runtime.h>
#include <hip/hip_bf16.h>
#include <math.h>

// ---------------- types ----------------
typedef __bf16 bf16;
typedef bf16  bf16x16 __attribute__((ext_vector_type(16)));
typedef float f32x8   __attribute__((ext_vector_type(8)));
typedef unsigned uint32x4 __attribute__((ext_vector_type(4)));
typedef int int32x4 __attribute__((ext_vector_type(4)));
typedef int int32x8 __attribute__((ext_vector_type(8)));

union FragB   { bf16x16 v; uint4 u[2]; };   // 32 bytes: 16 bf16 per lane
union FragAcc { f32x8 v; float f[8]; };
union PackBF  { bf16 h[2]; unsigned u; unsigned short s[2]; };

__device__ inline f32x8 zero8() {
  f32x8 z = {0.f,0.f,0.f,0.f,0.f,0.f,0.f,0.f};
  return z;
}

__device__ inline f32x8 wmma_bf16(bf16x16 a, bf16x16 b, f32x8 c) {
  return __builtin_amdgcn_wmma_f32_16x16x32_bf16(false, a, false, b, (short)0, c, false, false);
}

__device__ inline unsigned pack_bf16(float lo, float hi) {
  PackBF p; p.h[0] = (bf16)lo; p.h[1] = (bf16)hi; return p.u;
}

// Problem constants
#define BB 8
#define NN 4096
#define CC 1024
#define HH 16
#define DH 64
#define C3 3072
#define MM (BB*NN)   // 32768

// =====================================================================
// TDM: load a 128x32 bf16 tile (row stride `srcStride` elems) into LDS
// rows padded to 80B (64B data + 16B pad) == [.][40] bf16 arrays.
// D# per CDNA5 ISA. Issued once per wave.
// Toolchain here exposes the 6-arg builtin (extra zero group before cpol).
// =====================================================================
__device__ inline void tdm_load_tile_128x32(const bf16* gsrc, unsigned lds_off,
                                            unsigned srcStride) {
  unsigned long long ga = (unsigned long long)(uintptr_t)gsrc;
  uint32x4 g0;
  g0[0] = 1u;                                   // count=1, user descriptor
  g0[1] = lds_off;                              // lds_addr [63:32]
  g0[2] = (unsigned)(ga & 0xffffffffu);         // global_addr lo
  g0[3] = (unsigned)((ga >> 32) & 0x01ffffffu) | (2u << 30);  // addr hi | type=2
  int32x8 g1;
  // data_size=1(2B)<<16 | pad_enable<<20 | pad_interval=3(16 DW)<<22 | pad_amount=3(4 DW)<<25
  g1[0] = (1 << 16) | (1 << 20) | (3 << 22) | (3 << 25);
  g1[1] = (int)((srcStride & 0xffffu) << 16);   // tensor_dim0[15:0] @ [63:48]
  g1[2] = (int)(((srcStride >> 16) & 0xffffu)   // tensor_dim0[31:16] @ [79:64]
              | (0x8000u << 16));               // tensor_dim1[15:0]=32768 @ [95:80]
  g1[3] = (int)(32u << 16);                     // tensor_dim1 hi=0 | tile_dim0=32 @[127:112]
  g1[4] = 128;                                  // tile_dim1=128 | tile_dim2=0
  g1[5] = (int)srcStride;                       // tensor_dim0_stride[31:0]
  g1[6] = 0;                                    // stride hi | tensor_dim1_stride lo
  g1[7] = 0;
  int32x4 gz4 = {0, 0, 0, 0};
  int32x8 gz8 = {0, 0, 0, 0, 0, 0, 0, 0};
  __builtin_amdgcn_tensor_load_to_lds(g0, g1, gz4, gz4, gz8, 0);
}

// =====================================================================
// Kernel 1: qkv = x @ w_qkv   (fp32 in -> bf16 out, split layouts)
//   q,k -> [B*H*64][4096] (transposed)    v -> [B*N][1024] (natural)
// BM=128, BN=256, BK=32; 8 waves, 64x64 per wave (16 WMMA / K-step / wave)
// Software-pipelined: tile k+1 staged in regs during compute of tile k.
// =====================================================================
__global__ __launch_bounds__(256) void k_qkv(const float* __restrict__ X,
                                             const float* __restrict__ W,
                                             bf16* __restrict__ Q,
                                             bf16* __restrict__ K,
                                             bf16* __restrict__ V) {
  __shared__ __align__(16) bf16 As[128][40];   // [m][k] 80B rows (conflict-free)
  __shared__ __align__(16) bf16 Bs[256][40];   // transposed [n][k]

  const int tid  = threadIdx.x;
  const int lane = tid & 31;
  const int wave = tid >> 5;
  const int m0 = blockIdx.x * 128;
  const int n0 = blockIdx.y * 256;
  const int wm = (wave & 1) * 64;
  const int wn = (wave >> 1) * 64;

  FragAcc acc[4][4];
#pragma unroll
  for (int i = 0; i < 4; i++)
#pragma unroll
    for (int j = 0; j < 4; j++) acc[i][j].v = zero8();

  const float* xA = X + (size_t)m0 * CC;
  const float* wB = W + n0;

  // staging assignments
  const int ar = tid >> 1, ac = (tid & 1) * 16;        // A: 16 f32 per thread
  const int bn = (tid & 15) * 16, bk = (tid >> 4) * 2; // B: rows bk,bk+1 x 16 cols

  unsigned areg[8];   // 16 bf16 (k-contiguous pairs)
  unsigned breg[16];  // 16 cols x (2 adjacent k) packed

  // ---- stage tile kk into registers (with f32->bf16 cvt) ----
  auto stageA = [&](int kk) {
    const float4* s = (const float4*)(xA + (size_t)ar * CC + kk + ac);
    float4 f0 = s[0], f1 = s[1], f2 = s[2], f3 = s[3];
    areg[0] = pack_bf16(f0.x, f0.y); areg[1] = pack_bf16(f0.z, f0.w);
    areg[2] = pack_bf16(f1.x, f1.y); areg[3] = pack_bf16(f1.z, f1.w);
    areg[4] = pack_bf16(f2.x, f2.y); areg[5] = pack_bf16(f2.z, f2.w);
    areg[6] = pack_bf16(f3.x, f3.y); areg[7] = pack_bf16(f3.z, f3.w);
  };
  auto stageB = [&](int kk) {
    const float4* s0 = (const float4*)(wB + (size_t)(kk + bk) * C3 + bn);
    const float4* s1 = (const float4*)(wB + (size_t)(kk + bk + 1) * C3 + bn);
    float4 a0 = s0[0], a1 = s0[1], a2 = s0[2], a3 = s0[3];
    float4 c0 = s1[0], c1 = s1[1], c2 = s1[2], c3 = s1[3];
    float lo[16] = {a0.x,a0.y,a0.z,a0.w, a1.x,a1.y,a1.z,a1.w,
                    a2.x,a2.y,a2.z,a2.w, a3.x,a3.y,a3.z,a3.w};
    float hi[16] = {c0.x,c0.y,c0.z,c0.w, c1.x,c1.y,c1.z,c1.w,
                    c2.x,c2.y,c2.z,c2.w, c3.x,c3.y,c3.z,c3.w};
#pragma unroll
    for (int j = 0; j < 16; j++) breg[j] = pack_bf16(lo[j], hi[j]);
  };

  stageA(0); stageB(0);

  const int NSTEP = CC / 32;
  for (int i = 0; i < NSTEP; i++) {
    if (i) __syncthreads();
    // commit regs -> LDS
    {
      uint4* d = (uint4*)&As[ar][ac];
      d[0] = *(uint4*)&areg[0]; d[1] = *(uint4*)&areg[4];
#pragma unroll
      for (int j = 0; j < 16; j++)
        *(unsigned*)&Bs[bn + j][bk] = breg[j];   // b32 stores (2 k per store)
    }
    if (i + 1 < NSTEP) { stageA((i + 1) * 32); stageB((i + 1) * 32); }
    __syncthreads();

    // ---- 16 WMMA per wave ----
    const int koff = (lane >> 4) * 8;
    const int kb   = (lane >> 4) * 8;
    FragB a[4], b[4];
#pragma unroll
    for (int mi = 0; mi < 4; mi++) {
      int r = wm + mi * 16 + (lane & 15);
      a[mi].u[0] = *(const uint4*)&As[r][koff];
      a[mi].u[1] = *(const uint4*)&As[r][koff + 16];
    }
#pragma unroll
    for (int nj = 0; nj < 4; nj++) {
      int nc = wn + nj * 16 + (lane & 15);
      b[nj].u[0] = *(const uint4*)&Bs[nc][kb];
      b[nj].u[1] = *(const uint4*)&Bs[nc][kb + 16];
    }
#pragma unroll
    for (int mi = 0; mi < 4; mi++)
#pragma unroll
      for (int nj = 0; nj < 4; nj++)
        acc[mi][nj].v = wmma_bf16(a[mi].v, b[nj].v, acc[mi][nj].v);
  }

  // ---- epilogue: q/k transposed, v natural (region uniform per block) ----
#pragma unroll
  for (int mi = 0; mi < 4; mi++) {
    int mg0 = m0 + wm + mi * 16 + 8 * (lane >> 4);
#pragma unroll
    for (int nj = 0; nj < 4; nj++) {
      int cg = n0 + wn + nj * 16 + (lane & 15);
      if (cg < 2048) {
        bf16* dst = (cg < CC) ? Q : K;
        int c = cg & (CC - 1);
        int bidx = mg0 >> 12;
        int tok  = mg0 & (NN - 1);
        size_t base = (size_t)(bidx * CC + c) * NN + tok;
        bf16 t[8];
#pragma unroll
        for (int r = 0; r < 8; r++) t[r] = (bf16)acc[mi][nj].f[r];
        *(uint4*)(dst + base) = *(uint4*)t;       // 8 consecutive tokens
      } else {
        int c = cg - 2048;
#pragma unroll
        for (int r = 0; r < 8; r++)
          V[(size_t)(mg0 + r) * CC + c] = (bf16)acc[mi][nj].f[r];
      }
    }
  }
}

// =====================================================================
// Kernel 2: inverse L2 norms of q,k rows (4096 bf16 each)
// =====================================================================
__global__ __launch_bounds__(256) void k_norm(const bf16* __restrict__ Q,
                                              const bf16* __restrict__ Kk,
                                              float* __restrict__ invq,
                                              float* __restrict__ invk) {
  const int row = blockIdx.x;
  const bf16* src;
  float* dst;
  if (row < BB * HH * DH) { src = Q  + (size_t)row * NN;              dst = invq + row; }
  else                    { src = Kk + (size_t)(row - BB*HH*DH) * NN; dst = invk + (row - BB*HH*DH); }

  union { uint4 u; bf16 h[8]; } pk;
  float s = 0.f;
  for (int i = threadIdx.x; i < NN / 8; i += 256) {
    pk.u = ((const uint4*)src)[i];
#pragma unroll
    for (int j = 0; j < 8; j++) { float v = (float)pk.h[j]; s += v * v; }
  }
  __shared__ float red[256];
  red[threadIdx.x] = s;
  __syncthreads();
  for (int step = 128; step > 0; step >>= 1) {
    if (threadIdx.x < step) red[threadIdx.x] += red[threadIdx.x + step];
    __syncthreads();
  }
  if (threadIdx.x == 0) *dst = 1.0f / fmaxf(sqrtf(red[0]), 1e-12f);
}

// =====================================================================
// Kernel 3: per-(b,h) channel attention (Gram + softmax + attn@v)
// =====================================================================
__global__ __launch_bounds__(256) void k_attn(const bf16* __restrict__ Q,
                                              const bf16* __restrict__ Kk,
                                              const bf16* __restrict__ V,
                                              const float* __restrict__ invq,
                                              const float* __restrict__ invk,
                                              const float* __restrict__ temp,
                                              bf16* __restrict__ AO) {
  const int bh = blockIdx.x;
  const int b = bh >> 4, h = bh & 15;
  const int tid = threadIdx.x, lane = tid & 31, wave = tid >> 5;

  __shared__ float Gs[64][68];
  __shared__ __align__(16) bf16 Att[64][72];

  const bf16* qrow = Q  + (size_t)bh * DH * NN;
  const bf16* krow = Kk + (size_t)bh * DH * NN;
  const int mt = wave & 3;
  const int ntb = (wave >> 2) * 2;

  FragAcc g0, g1;
  g0.v = zero8(); g1.v = zero8();

  const bf16* aptr  = qrow + (size_t)(mt * 16 + (lane & 15)) * NN + (lane >> 4) * 16;
  const bf16* bptr0 = krow + (size_t)(ntb * 16 + (lane & 15)) * NN + (lane >> 4) * 16;
  const bf16* bptr1 = bptr0 + (size_t)16 * NN;

  for (int kk = 0; kk < NN; kk += 32) {
    FragB a, c0, c1;
    a.u[0]  = *(const uint4*)(aptr + kk);
    a.u[1]  = *(const uint4*)(aptr + kk + 8);
    c0.u[0] = *(const uint4*)(bptr0 + kk);
    c0.u[1] = *(const uint4*)(bptr0 + kk + 8);
    c1.u[0] = *(const uint4*)(bptr1 + kk);
    c1.u[1] = *(const uint4*)(bptr1 + kk + 8);
    if (kk + 512 < NN) {                 // stream-ahead hint (global_prefetch)
      __builtin_prefetch(aptr + kk + 512, 0, 1);
    }
    g0.v = wmma_bf16(a.v, c0.v, g0.v);
    g1.v = wmma_bf16(a.v, c1.v, g1.v);
  }

  {
    const float ts = temp[h];
    const float* iq = invq + bh * DH;
    const float* ik = invk + bh * DH;
#pragma unroll
    for (int j = 0; j < 2; j++) {
      FragAcc* g = (j == 0) ? &g0 : &g1;
      int e = (ntb + j) * 16 + (lane & 15);
      float se = ik[e] * ts;
#pragma unroll
      for (int r = 0; r < 8; r++) {
        int d = mt * 16 + r + 8 * (lane >> 4);
        Gs[d][e] = g->f[r] * iq[d] * se;
      }
    }
  }
  __syncthreads();

  if (tid < 64) {
    float mx = -3.4e38f;
    for (int e = 0; e < 64; e++) mx = fmaxf(mx, Gs[tid][e]);
    float sum = 0.f;
    for (int e = 0; e < 64; e++) { float ex = __expf(Gs[tid][e] - mx); Gs[tid][e] = ex; sum += ex; }
    float inv = 1.0f / sum;
    for (int e = 0; e < 64; e++) Att[tid][e] = (bf16)(Gs[tid][e] * inv);
  }
  __syncthreads();

  FragB bfr[2][4];
#pragma unroll
  for (int ks = 0; ks < 2; ks++)
#pragma unroll
    for (int nt = 0; nt < 4; nt++) {
      int d  = nt * 16 + (lane & 15);
      int kb = ks * 32 + (lane >> 4) * 8;
      bfr[ks][nt].u[0] = *(const uint4*)&Att[d][kb];
      bfr[ks][nt].u[1] = *(const uint4*)&Att[d][kb + 16];
    }

  const bf16* vbase = V  + (size_t)b * NN * CC + h * DH;
  bf16*       obase = AO + (size_t)b * NN * CC + h * DH;

  for (int mtile = wave; mtile < NN / 16; mtile += 8) {
    int tok0 = mtile * 16;
    const bf16* ap = vbase + (size_t)(tok0 + (lane & 15)) * CC + (lane >> 4) * 8;
    FragB a0, a1;
    a0.u[0] = *(const uint4*)(ap);
    a0.u[1] = *(const uint4*)(ap + 16);
    a1.u[0] = *(const uint4*)(ap + 32);
    a1.u[1] = *(const uint4*)(ap + 48);

    FragAcc o[4];
#pragma unroll
    for (int nt = 0; nt < 4; nt++) {
      o[nt].v = zero8();
      o[nt].v = wmma_bf16(a0.v, bfr[0][nt].v, o[nt].v);
      o[nt].v = wmma_bf16(a1.v, bfr[1][nt].v, o[nt].v);
    }
#pragma unroll
    for (int nt = 0; nt < 4; nt++) {
      int cg = nt * 16 + (lane & 15);
#pragma unroll
      for (int r = 0; r < 8; r++) {
        int tok = tok0 + r + 8 * (lane >> 4);
        obase[(size_t)tok * CC + cg] = (bf16)o[nt].f[r];
      }
    }
  }
}

// =====================================================================
// Kernel 4: out = attn_out @ w_proj + b_proj
// A (bf16) staged by the Tensor Data Mover into double-buffered padded LDS;
// TDM for tile i+1 overlaps compute of tile i (s_wait_tensorcnt pipeline).
// BM=128, BN=256; 8 waves, 64x64 per wave.
// =====================================================================
__global__ __launch_bounds__(256) void k_proj(const bf16* __restrict__ A,
                                              const float* __restrict__ W,
                                              const float* __restrict__ bias,
                                              float* __restrict__ Out) {
  __shared__ __align__(16) bf16 Ab[2][128][40];   // TDM targets (80B rows)
  __shared__ __align__(16) bf16 Bs[256][40];      // transposed [n][k]

  const int tid  = threadIdx.x;
  const int lane = tid & 31;
  const int wave = tid >> 5;
  const int m0 = blockIdx.x * 128;
  const int n0 = blockIdx.y * 256;
  const int wm = (wave & 1) * 64;
  const int wn = (wave >> 1) * 64;

  FragAcc acc[4][4];
#pragma unroll
  for (int i = 0; i < 4; i++)
#pragma unroll
    for (int j = 0; j < 4; j++) acc[i][j].v = zero8();

  const bf16* aT = A + (size_t)m0 * CC;
  const float* wB = W + n0;

  const int bn = (tid & 15) * 16, bk = (tid >> 4) * 2;
  unsigned breg[16];
  auto stageB = [&](int kk) {
    const float4* s0 = (const float4*)(wB + (size_t)(kk + bk) * CC + bn);
    const float4* s1 = (const float4*)(wB + (size_t)(kk + bk + 1) * CC + bn);
    float4 a0 = s0[0], a1 = s0[1], a2 = s0[2], a3 = s0[3];
    float4 c0 = s1[0], c1 = s1[1], c2 = s1[2], c3 = s1[3];
    float lo[16] = {a0.x,a0.y,a0.z,a0.w, a1.x,a1.y,a1.z,a1.w,
                    a2.x,a2.y,a2.z,a2.w, a3.x,a3.y,a3.z,a3.w};
    float hi[16] = {c0.x,c0.y,c0.z,c0.w, c1.x,c1.y,c1.z,c1.w,
                    c2.x,c2.y,c2.z,c2.w, c3.x,c3.y,c3.z,c3.w};
#pragma unroll
    for (int j = 0; j < 16; j++) breg[j] = pack_bf16(lo[j], hi[j]);
  };

  const unsigned lds_a0 = (unsigned)(uintptr_t)&Ab[0][0][0];
  const unsigned lds_a1 = (unsigned)(uintptr_t)&Ab[1][0][0];

  // prologue: TDM tile 0, B tile 0 into regs
  if (wave == 0) tdm_load_tile_128x32(aT, lds_a0, CC);
  stageB(0);

  const int NSTEP = CC / 32;   // 32
  for (int i = 0; i < NSTEP; i++) {
    if (i) __syncthreads();    // prior compute done: Bs and Ab[(i+1)&1] free
    if (wave == 0 && i + 1 < NSTEP)
      tdm_load_tile_128x32(aT + (i + 1) * 32, (i & 1) ? lds_a0 : lds_a1, CC);
    {
#pragma unroll
      for (int j = 0; j < 16; j++)
        *(unsigned*)&Bs[bn + j][bk] = breg[j];
    }
    if (i + 1 < NSTEP) stageB((i + 1) * 32);
    if (wave == 0) {
      if (i + 1 < NSTEP) __builtin_amdgcn_s_wait_tensorcnt(1);  // tile i landed
      else               __builtin_amdgcn_s_wait_tensorcnt(0);
    }
    __syncthreads();

    const int koff = (lane >> 4) * 8;
    const int kb   = (lane >> 4) * 8;
    FragB a[4], b[4];
#pragma unroll
    for (int mi = 0; mi < 4; mi++) {
      int r = wm + mi * 16 + (lane & 15);
      a[mi].u[0] = *(const uint4*)&Ab[i & 1][r][koff];
      a[mi].u[1] = *(const uint4*)&Ab[i & 1][r][koff + 16];
    }
#pragma unroll
    for (int nj = 0; nj < 4; nj++) {
      int nc = wn + nj * 16 + (lane & 15);
      b[nj].u[0] = *(const uint4*)&Bs[nc][kb];
      b[nj].u[1] = *(const uint4*)&Bs[nc][kb + 16];
    }
#pragma unroll
    for (int mi = 0; mi < 4; mi++)
#pragma unroll
      for (int nj = 0; nj < 4; nj++)
        acc[mi][nj].v = wmma_bf16(a[mi].v, b[nj].v, acc[mi][nj].v);
  }

  // epilogue: bias add, fp32 store (coalesced across lanes 0-15)
#pragma unroll
  for (int mi = 0; mi < 4; mi++) {
    int mg0 = m0 + wm + mi * 16 + 8 * (lane >> 4);
#pragma unroll
    for (int nj = 0; nj < 4; nj++) {
      int cg = n0 + wn + nj * 16 + (lane & 15);
      float bv = bias[cg];
#pragma unroll
      for (int r = 0; r < 8; r++)
        Out[(size_t)(mg0 + r) * CC + cg] = acc[mi][nj].f[r] + bv;
    }
  }
}

// =====================================================================
extern "C" void kernel_launch(void* const* d_in, const int* in_sizes, int n_in,
                              void* d_out, int out_size, void* d_ws, size_t ws_size,
                              hipStream_t stream) {
  const float* x     = (const float*)d_in[0];
  const float* wqkv  = (const float*)d_in[1];
  const float* temp  = (const float*)d_in[2];
  const float* wproj = (const float*)d_in[3];
  const float* bproj = (const float*)d_in[4];
  float* out = (float*)d_out;

  const size_t SZ = (size_t)BB * HH * DH * NN * sizeof(bf16);  // 64 MB
  char* ws = (char*)d_ws;
  bf16* Q    = (bf16*)(ws);
  bf16* K    = (bf16*)(ws + SZ);
  bf16* V    = (bf16*)(ws + 2 * SZ);
  bf16* AO   = (bf16*)(ws + 3 * SZ);
  float* invq = (float*)(ws + 4 * SZ);
  float* invk = invq + BB * HH * DH;

  dim3 g1(MM / 128, C3 / 256);   // 256 x 12
  k_qkv<<<g1, 256, 0, stream>>>(x, wqkv, Q, K, V);

  k_norm<<<2 * BB * HH * DH, 256, 0, stream>>>(Q, K, invq, invk);

  k_attn<<<BB * HH, 256, 0, stream>>>(Q, K, V, invq, invk, temp, AO);

  dim3 g4(MM / 128, CC / 256);   // 256 x 4
  k_proj<<<g4, 256, 0, stream>>>(AO, wproj, bproj, out);
}
